// VectorQuantizedVAE_48816598286642
// MI455X (gfx1250) — matrix operand order repaired and checked by reading, not compiled, not run on hardware
//
#include <hip/hip_runtime.h>
#include <hip/hip_bf16.h>
#include <math.h>
#include <stdint.h>

// ---------------------------------------------------------------------------
// VQ-VAE forward on MI455X (gfx1250, wave32, WMMA).
// All convolutions (incl. transposed via lhs-dilation) run through one
// implicit-GEMM kernel using V_WMMA_F32_16X16X32_BF16 (bf16 in, f32 accum).
// Activations fp32 NHWC with Cin % 8 == 0 (x is channel-padded 3->8);
// weights packed to bf16 [CoutPad][taps*CinPad] so the B tile streams via
// GLOBAL_LOAD_ASYNC_TO_LDS_B128 (ASYNCcnt) with no bounds branch.
// K-order is (tap, channel): the hot loop has no integer division and the
// A gather is either an aligned float4x2 load or pure zero-fill.
// ---------------------------------------------------------------------------

typedef __bf16 bf16_t;
typedef __attribute__((ext_vector_type(8)))  __bf16 v8bf;
typedef __attribute__((ext_vector_type(16))) __bf16 v16bf;
typedef __attribute__((ext_vector_type(8)))  float  v8f;

#define BM 64
#define BN 64
#define LDS_STRIDE 40   // 32 bf16 + 8 pad per row -> 80B rows, 16B aligned chunks
#define TPB 256
#define EPSV 1e-5f

struct ConvDesc {
  int N, IH, IW, Cin, CinPad;
  int OH, OW, Cout;
  int kh, kw, stride, pad, lhsDil;
  int KdPad;     // kh*kw*CinPad
  long M;        // N*OH*OW
  int reluIn;    // apply ReLU to input activations while gathering
};

// ---------------------------------------------------------------------------
// Implicit-GEMM conv: out[m][co] = sum_k im2col(in)[m][k] * wBT[co][k] + bias
// k = (r*kw + s)*CinPad + c (channel fastest -> contiguous NHWC reads).
// lhsDil>1 models ConvTranspose as a conv over the zero-dilated input.
// REQUIRES: Cin % 8 == 0 (activation buffers are channel-padded).
// ---------------------------------------------------------------------------
__global__ __launch_bounds__(TPB)
void conv_implicit_wmma(const float* __restrict__ in,
                        const bf16_t* __restrict__ wBT,
                        const float* __restrict__ bias,
                        float* __restrict__ out, ConvDesc d)
{
  __shared__ __align__(16) bf16_t As[BM * LDS_STRIDE];
  __shared__ __align__(16) bf16_t Bs[BN * LDS_STRIDE];

  const int tid   = threadIdx.x;
  const int lane  = tid & 31;
  const int wave  = tid >> 5;
  const int waveM = wave & 3;          // 4 waves along M (4*16 = 64)
  const int waveN = wave >> 2;         // 2 waves along N (2*32 = 64)
  const long mBase = (long)blockIdx.x * BM;
  const int  nBase = blockIdx.y * BN;

  v8f acc0 = {};
  v8f acc1 = {};

  // ---- per-thread prologue (the only place with divisions) ----
  const int arow = tid >> 2;            // 0..63 : A row this thread fills
  const int akc  = (tid & 3) * 8;       // 0,8,16,24 : channel sub-chunk
  const long m   = mBase + arow;
  const int ohow = d.OH * d.OW;
  int n_ = 0, oh_ = 0, ow_ = 0;
  const bool mvalid = (m < d.M);
  if (mvalid) {
    n_ = (int)(m / ohow);
    int rem = (int)(m % ohow);
    oh_ = rem / d.OW;
    ow_ = rem % d.OW;
  }
  const int thBase = oh_ * d.stride - d.pad;
  const int twBase = ow_ * d.stride - d.pad;
  const float* nPtr = in + (long)n_ * d.IH * d.IW * d.Cin;
  // branchless ReLU bound: fmax(v, -inf) is identity
  const float lo = d.reluIn ? 0.0f : -__builtin_inff();

  const int brow = tid >> 2;            // 0..63 : B row (output channel)
  const int bkc  = (tid & 3) * 8;       // K sub-chunk
  // LDS byte offset of this thread's B destination (low 32 bits of the flat
  // shared-memory address are the LDS offset on gfx12 apertures).
  const unsigned ldsB = (unsigned)(uintptr_t)&Bs[brow * LDS_STRIDE + bkc];
  const long wRow = (long)(nBase + brow) * d.KdPad;   // element offset of B row

  const int taps = d.kh * d.kw;
  const int CinPad = d.CinPad;

  int r = 0, s = 0;
  for (int t = 0; t < taps; ++t) {
    // ---- spatial resolve, once per tap ----
    bool sval = false;
    const float* rowPtr = nullptr;
    {
      int th = thBase + r, tw = twBase + s;
      if (mvalid && th >= 0 && tw >= 0) {
        if (d.lhsDil == 1) {
          if (th < d.IH && tw < d.IW)
            { sval = true; rowPtr = nPtr + ((long)th * d.IW + tw) * d.Cin; }
        } else if ((th % d.lhsDil) == 0 && (tw % d.lhsDil) == 0) {
          int ih = th / d.lhsDil, iw = tw / d.lhsDil;
          if (ih < d.IH && iw < d.IW)
            { sval = true; rowPtr = nPtr + ((long)ih * d.IW + iw) * d.Cin; }
        }
      }
    }
    if (sval) __builtin_prefetch(rowPtr + akc, 0, 0);   // global_prefetch_b8
    const long kTap = (long)t * CinPad;

    for (int cb = 0; cb < CinPad; cb += 32) {
      // ---- B tile chunk: async copy global -> LDS (16B/thread) ----
      {
        unsigned goff = (unsigned)((wRow + kTap + cb + bkc) * 2);
        asm volatile("global_load_async_to_lds_b128 %0, %1, %2"
                     :: "v"(ldsB), "v"(goff), "s"(wBT) : "memory");
      }
      // ---- A tile chunk: aligned float4x2 load or pure zero-fill ----
      {
        const int c0 = cb + akc;
        v8bf av = {};
        if (sval && (c0 + 8) <= d.Cin) {
          float4 f0 = *(const float4*)(rowPtr + c0);
          float4 f1 = *(const float4*)(rowPtr + c0 + 4);
          float fv[8] = {f0.x, f0.y, f0.z, f0.w, f1.x, f1.y, f1.z, f1.w};
          #pragma unroll
          for (int j = 0; j < 8; ++j) av[j] = (bf16_t)fmaxf(fv[j], lo);
        }
        *(v8bf*)&As[arow * LDS_STRIDE + akc] = av;
      }
      // async B copy must land in LDS before anyone reads the tile
      asm volatile("s_wait_asynccnt 0x0" ::: "memory");
      __syncthreads();

      // ---- fragments in contiguous k order (k = 16h + s), same for A and B ----
      // A 16x32: lanes 0-15 hold K 0..15; lanes 16-31 K 16..31
      const int aRow = waveM * 16 + (lane & 15);
      const int aK0  = (lane < 16) ? 0 : 16;
      v8bf a0 = *(const v8bf*)&As[aRow * LDS_STRIDE + aK0];
      v8bf a1 = *(const v8bf*)&As[aRow * LDS_STRIDE + aK0 + 8];
      v16bf afrag = __builtin_shufflevector(a0, a1,
          0,1,2,3,4,5,6,7,8,9,10,11,12,13,14,15);

      // B 32x16: lane holds col N=lane%16, K 0..15 (lanes<16) / 16..31
      const int bCol = waveN * 32 + (lane & 15);
      const int bK0  = (lane < 16) ? 0 : 16;
      v8bf b00 = *(const v8bf*)&Bs[bCol * LDS_STRIDE + bK0];
      v8bf b01 = *(const v8bf*)&Bs[bCol * LDS_STRIDE + bK0 + 8];
      v16bf bfrag0 = __builtin_shufflevector(b00, b01,
          0,1,2,3,4,5,6,7,8,9,10,11,12,13,14,15);
      v8bf b10 = *(const v8bf*)&Bs[(bCol + 16) * LDS_STRIDE + bK0];
      v8bf b11 = *(const v8bf*)&Bs[(bCol + 16) * LDS_STRIDE + bK0 + 8];
      v16bf bfrag1 = __builtin_shufflevector(b10, b11,
          0,1,2,3,4,5,6,7,8,9,10,11,12,13,14,15);

      acc0 = __builtin_amdgcn_wmma_f32_16x16x32_bf16(false, afrag, false, bfrag0,
                                                     (short)0, acc0, false, false);
      acc1 = __builtin_amdgcn_wmma_f32_16x16x32_bf16(false, afrag, false, bfrag1,
                                                     (short)0, acc1, false, false);
      __syncthreads();
    }
    if (++s == d.kw) { s = 0; ++r; }
  }

  // ---- epilogue: C/D layout: VGPR v -> M = v + 8*(lane>=16), N = lane%16 ----
  const int colLane = lane & 15;
  const int mOff    = (lane < 16) ? 0 : 8;
  const int co0 = nBase + waveN * 32 + colLane;
  const int co1 = co0 + 16;
  const float bias0 = (bias && co0 < d.Cout) ? bias[co0] : 0.0f;
  const float bias1 = (bias && co1 < d.Cout) ? bias[co1] : 0.0f;
  #pragma unroll
  for (int v = 0; v < 8; ++v) {
    long mm = mBase + waveM * 16 + mOff + v;
    if (mm < d.M) {
      if (co0 < d.Cout) out[mm * d.Cout + co0] = acc0[v] + bias0;
      if (co1 < d.Cout) out[mm * d.Cout + co1] = acc1[v] + bias1;
    }
  }
}

// ---------------------------------------------------------------------------
// Weight packing: fp32 -> bf16 [CoutPad][kh*kw*CinPad], k=(r*kw+s)*CinPad+c,
// zero padded in both channel dims so the GEMM needs no bounds checks on B.
// wCin = real channel count in the raw weight tensor (may be < Cin when the
// activation buffer is channel-padded). tmode=1: ConvTranspose (I,O,kh,kw)
// weight -> spatially flipped, I/O swapped.
// ---------------------------------------------------------------------------
__global__ void pack_weight(const float* __restrict__ w, bf16_t* __restrict__ bt,
                            int Cout, int CoutPad, int wCin, int CinPad,
                            int kh, int kw, int tmode)
{
  long KdPad = (long)kh * kw * CinPad;
  long total = (long)CoutPad * KdPad;
  long i = (long)blockIdx.x * blockDim.x + threadIdx.x;
  if (i >= total) return;
  int o = (int)(i / KdPad);
  int k = (int)(i % KdPad);
  int c  = k % CinPad;
  int rs = k / CinPad;
  int s  = rs % kw;
  int r  = rs / kw;
  float v = 0.0f;
  if (o < Cout && c < wCin) {
    if (!tmode) v = w[(((long)o * wCin + c) * kh + r) * kw + s];
    else        v = w[(((long)c * Cout + o) * kh + (kh - 1 - r)) * kw + (kw - 1 - s)];
  }
  bt[i] = (bf16_t)v;
}

// ---------------------------------------------------------------------------
// Training-mode BN (biased variance over all rows per channel), NHWC.
// ---------------------------------------------------------------------------
__global__ void bn_stats(const float* __restrict__ in, float* __restrict__ mean,
                         float* __restrict__ invstd, long M, int C, float eps)
{
  int c = blockIdx.x;
  float s = 0.0f, s2 = 0.0f;
  for (long m = threadIdx.x; m < M; m += blockDim.x) {
    float v = in[m * C + c];
    s += v; s2 += v * v;
  }
  __shared__ float sh[TPB], sh2[TPB];
  sh[threadIdx.x] = s; sh2[threadIdx.x] = s2;
  __syncthreads();
  for (int o = TPB / 2; o > 0; o >>= 1) {
    if ((int)threadIdx.x < o) {
      sh[threadIdx.x]  += sh[threadIdx.x + o];
      sh2[threadIdx.x] += sh2[threadIdx.x + o];
    }
    __syncthreads();
  }
  if (threadIdx.x == 0) {
    float mu  = sh[0] / (float)M;
    float var = sh2[0] / (float)M - mu * mu;
    mean[c]   = mu;
    invstd[c] = rsqrtf(var + eps);
  }
}

__global__ void bn_apply(const float* __restrict__ in, const float* __restrict__ res,
                         float* __restrict__ out, const float* __restrict__ mean,
                         const float* __restrict__ invstd, const float* __restrict__ g,
                         const float* __restrict__ b, long total, int C, int relu)
{
  long i = (long)blockIdx.x * blockDim.x + threadIdx.x;
  if (i >= total) return;
  int c = (int)(i % C);
  float y = (in[i] - mean[c]) * invstd[c] * g[c] + b[c];
  if (relu) y = fmaxf(y, 0.0f);
  if (res)  y += res[i];
  out[i] = y;
}

// ---------------------------------------------------------------------------
// Layout shuffles.
// ---------------------------------------------------------------------------
// NCHW -> NHWC with channel padding to Cpad (zeros in padded channels).
__global__ void nchw_to_nhwc_pad(const float* __restrict__ in, float* __restrict__ out,
                                 int N, int C, int H, int W, int Cpad)
{
  long total = (long)N * H * W * Cpad;
  long i = (long)blockIdx.x * blockDim.x + threadIdx.x;
  if (i >= total) return;
  int c = (int)(i % Cpad); long r = i / Cpad;
  int w = (int)(r % W); r /= W;
  int h = (int)(r % H); int n = (int)(r / H);
  out[i] = (c < C) ? in[(((long)n * C + c) * H + h) * W + w] : 0.0f;
}

__global__ void nhwc_to_nchw(const float* __restrict__ in, float* __restrict__ out,
                             int N, int H, int W, int C, int doTanh)
{
  long total = (long)N * H * W * C;
  long i = (long)blockIdx.x * blockDim.x + threadIdx.x;
  if (i >= total) return;
  int c = (int)(i % C); long r = i / C;
  int w = (int)(r % W); r /= W;
  int h = (int)(r % H); int n = (int)(r / H);
  float v = in[i];
  if (doTanh) v = tanhf(v);
  out[(((long)n * C + c) * H + h) * W + w] = v;
}

// ---------------------------------------------------------------------------
// Vector quantization helpers.
// ---------------------------------------------------------------------------
__global__ void enorm_kernel(const float* __restrict__ emb, float* __restrict__ en, int D)
{
  int j = blockIdx.x;
  __shared__ float sh[TPB];
  float s = 0.0f;
  for (int c = threadIdx.x; c < D; c += blockDim.x) {
    float v = emb[(long)j * D + c];
    s += v * v;
  }
  sh[threadIdx.x] = s;
  __syncthreads();
  for (int o = TPB / 2; o > 0; o >>= 1) {
    if ((int)threadIdx.x < o) sh[threadIdx.x] += sh[threadIdx.x + o];
    __syncthreads();
  }
  if (threadIdx.x == 0) en[j] = sh[0];
}

// one wave32 per row: argmin_j (||e_j||^2 - 2 * z.e_j)
__global__ void vq_argmin(const float* __restrict__ S, const float* __restrict__ en,
                          int* __restrict__ idx, long Mrows, int Kc)
{
  long row = (long)blockIdx.x * (blockDim.x >> 5) + (threadIdx.x >> 5);
  int lane = threadIdx.x & 31;
  if (row >= Mrows) return;
  float best = 3.402823e38f;
  int bj = 0;
  for (int j = lane; j < Kc; j += 32) {
    float dd = en[j] - 2.0f * S[row * Kc + j];
    if (dd < best || (dd == best && j < bj)) { best = dd; bj = j; }
  }
  for (int off = 16; off > 0; off >>= 1) {
    float ob = __shfl_xor(best, off, 32);
    int   oj = __shfl_xor(bj,   off, 32);
    if (ob < best || (ob == best && oj < bj)) { best = ob; bj = oj; }
  }
  if (lane == 0) idx[row] = bj;
}

__global__ void vq_gather(const int* __restrict__ idx, const float* __restrict__ emb,
                          float* __restrict__ zq, long Mrows, int D)
{
  long i = (long)blockIdx.x * blockDim.x + threadIdx.x;
  if (i >= Mrows * D) return;
  long m = i / D;
  int  c = (int)(i % D);
  zq[i] = emb[(long)idx[m] * D + c];
}

// ---------------------------------------------------------------------------
// Host orchestration.
// ---------------------------------------------------------------------------
static inline dim3 blocks1d(long total) {
  return dim3((unsigned)((total + TPB - 1) / TPB));
}

extern "C" void kernel_launch(void* const* d_in, const int* in_sizes, int n_in,
                              void* d_out, int out_size, void* d_ws, size_t ws_size,
                              hipStream_t stream)
{
  (void)in_sizes; (void)n_in; (void)out_size; (void)ws_size;

  auto P = [&](int i) { return (const float*)d_in[i]; };
  const float* x   = P(0);
  const float* emb = P(45);

  // workspace bump allocator
  size_t off = 0;
  auto alloc = [&](size_t bytes) {
    void* p = (char*)d_ws + off;
    off += (bytes + 255) & ~(size_t)255;
    return p;
  };
  float*  xn    = (float*)alloc(4194304UL * 4);          // x NHWC (32,128,128,8) padded
  float*  buf64 = (float*)alloc(33554432UL * 4);         // (32,64,64,256); also VQ scores
  float*  bufA  = (float*)alloc(8388608UL * 4);          // (32,32,32,256); also final NHWC out
  float*  bufB  = (float*)alloc(8388608UL * 4);
  float*  bufC  = (float*)alloc(8388608UL * 4);
  float*  bufD  = (float*)alloc(8388608UL * 4);          // z_q / decoder stream
  bf16_t* wp    = (bf16_t*)alloc(256UL * 4096 * 2);      // packed weights (max 2MB)
  float*  meanv = (float*)alloc(256 * 4);
  float*  invsd = (float*)alloc(256 * 4);
  float*  enr   = (float*)alloc(512 * 4);
  int*    idx   = (int*)alloc(32768 * 4);

  float* o_x  = (float*)d_out;                // x_tilde NCHW (32,3,128,128)
  float* o_ze = o_x + 1572864;                // z_e     NCHW (32,256,32,32)
  float* o_zq = o_ze + 8388608;               // z_q_bar NCHW (32,256,32,32)

  auto run_conv = [&](const float* inp, const float* wraw, const float* bias, float* outp,
                      int N, int IH, int IW, int Cin, int wCin,
                      int OH, int OW, int Cout,
                      int kh, int kw, int stride, int pad, int lhsDil,
                      int tmode, int reluIn) {
    ConvDesc d;
    d.N = N; d.IH = IH; d.IW = IW; d.Cin = Cin;
    d.CinPad = (Cin + 31) & ~31;
    d.OH = OH; d.OW = OW; d.Cout = Cout;
    d.kh = kh; d.kw = kw; d.stride = stride; d.pad = pad; d.lhsDil = lhsDil;
    d.KdPad = kh * kw * d.CinPad;
    d.M = (long)N * OH * OW;
    d.reluIn = reluIn;
    int CoutPad = (Cout + 63) & ~63;
    pack_weight<<<blocks1d((long)CoutPad * d.KdPad), TPB, 0, stream>>>(
        wraw, wp, Cout, CoutPad, wCin, d.CinPad, kh, kw, tmode);
    dim3 g((unsigned)((d.M + BM - 1) / BM), (unsigned)((Cout + BN - 1) / BN));
    conv_implicit_wmma<<<g, TPB, 0, stream>>>(inp, wp, bias, outp, d);
  };

  auto run_bn = [&](float* data, const float* res, float* outp,
                    const float* g, const float* b, long Mrows, int relu) {
    bn_stats<<<256, TPB, 0, stream>>>(data, meanv, invsd, Mrows, 256, EPSV);
    bn_apply<<<blocks1d(Mrows * 256), TPB, 0, stream>>>(
        data, res, outp, meanv, invsd, g, b, Mrows * 256, 256, relu);
  };

  auto resblock = [&](float* xbuf, int pb) {
    // h = conv1(relu(x)) ; bn1 ; relu ; h = conv2(h) ; bn2 ; x += h
    run_conv(xbuf, P(pb + 0), P(pb + 1), bufB,
             32, 32, 32, 256, 256, 32, 32, 256, 3, 3, 1, 1, 1, 0, /*reluIn=*/1);
    run_bn(bufB, nullptr, bufB, P(pb + 2), P(pb + 3), 32768, /*relu=*/1);
    run_conv(bufB, P(pb + 4), P(pb + 5), bufC,
             32, 32, 32, 256, 256, 32, 32, 256, 1, 1, 1, 0, 1, 0, 0);
    run_bn(bufC, xbuf, xbuf, P(pb + 6), P(pb + 7), 32768, 0);
  };

  // ---- encoder ----
  // x NCHW (32,3,128,128) -> NHWC padded to 8 channels (zeros in 3..7)
  nchw_to_nhwc_pad<<<blocks1d(4194304L), TPB, 0, stream>>>(x, xn, 32, 3, 128, 128, 8);
  // conv1 4x4 s2 p1: (32,128,128,8) -> (32,64,64,256), then BN+ReLU in-place
  run_conv(xn, P(1), P(2), buf64, 32, 128, 128, 8, 3, 64, 64, 256, 4, 4, 2, 1, 1, 0, 0);
  run_bn(buf64, nullptr, buf64, P(3), P(4), 131072, 1);
  // conv2 4x4 s2 p1: -> (32,32,32,256)
  run_conv(buf64, P(5), P(6), bufA, 32, 64, 64, 256, 256, 32, 32, 256, 4, 4, 2, 1, 1, 0, 0);
  // r1, r2
  resblock(bufA, 7);
  resblock(bufA, 15);
  // z_e output (NCHW)
  nhwc_to_nchw<<<blocks1d(8388608L), TPB, 0, stream>>>(bufA, o_ze, 32, 32, 32, 256, 0);

  // ---- vector quantization ----
  // scores = z_e @ emb^T as a 1x1 conv (Cout = 512 codes), scores in buf64
  run_conv(bufA, emb, nullptr, buf64, 32, 32, 32, 256, 256, 32, 32, 512, 1, 1, 1, 0, 1, 0, 0);
  enorm_kernel<<<512, TPB, 0, stream>>>(emb, enr, 256);
  vq_argmin<<<(unsigned)(32768 / (TPB / 32)), TPB, 0, stream>>>(buf64, enr, idx, 32768, 512);
  vq_gather<<<blocks1d(8388608L), TPB, 0, stream>>>(idx, emb, bufD, 32768, 256);
  // z_q_bar output (NCHW); straight-through forward value == z_q
  nhwc_to_nchw<<<blocks1d(8388608L), TPB, 0, stream>>>(bufD, o_zq, 32, 32, 32, 256, 0);

  // ---- decoder ----
  resblock(bufD, 23);  // r3
  resblock(bufD, 31);  // r4
  // relu -> ConvT dt1 (256->256, 4x4, s2, p1) == conv(lhsDil=2, pad=2) -> (32,64,64,256)
  run_conv(bufD, P(39), P(40), buf64, 32, 32, 32, 256, 256, 64, 64, 256,
           4, 4, 1, 2, 2, /*tmode=*/1, /*reluIn=*/1);
  run_bn(buf64, nullptr, buf64, P(41), P(42), 131072, 1);
  // ConvT dt2 (256->3) -> (32,128,128,3) NHWC in bufA
  run_conv(buf64, P(43), P(44), bufA, 32, 64, 64, 256, 256, 128, 128, 3,
           4, 4, 1, 2, 2, 1, 0);
  // tanh + NHWC->NCHW into x_tilde
  nhwc_to_nchw<<<blocks1d(1572864L), TPB, 0, stream>>>(bufA, o_x, 32, 128, 128, 3, 1);
}
